// MambaBlock_57793079935707
// MI455X (gfx1250) — matrix operand order repaired and checked by reading, not compiled, not run on hardware
//
#include <hip/hip_runtime.h>
#include <hip/hip_bf16.h>
#include <math.h>

// ---------------- CDNA5 vector types ----------------
typedef _Float16 v16h __attribute__((ext_vector_type(16)));
typedef _Float16 v8h  __attribute__((ext_vector_type(8)));
typedef float    v8f  __attribute__((ext_vector_type(8)));

// ---------------- problem constants -----------------
constexpr int BATCH   = 4;
constexpr int SEQ     = 4096;          // h*w = 64*64
constexpr int CH      = 128;
constexpr int DINNER  = 256;
constexpr int NHEADS  = 4;
constexpr int DSTATE  = 64;
constexpr int CONVCH  = 384;           // d_inner + 2*D_STATE
constexpr int NPROJ   = 644;           // 2*d_inner + 2*D_STATE + nheads
constexpr int NPROJ_PAD = 704;         // padded to multiple of 64 for WMMA tiling
constexpr int MROWS   = BATCH * SEQ;   // 16384
constexpr int KIM     = CH * 9;        // 1152 (im2col K for 3x3 conv)
constexpr int IMG     = 64;

// tile counts (N tiles of 16, K tiles of 32) -- all KT_* are even (pipeline needs it)
constexpr int NT_INPROJ = NPROJ_PAD / 16; // 44
constexpr int KT_INPROJ = CH / 32;        // 4
constexpr int NT_OUT    = CH / 16;        // 8
constexpr int KT_OUT    = DINNER / 32;    // 8
constexpr int NT_FF     = CH / 16;        // 8
constexpr int KT_FF     = KIM / 32;       // 36

// ---------------- workspace layout (bytes) ----------
constexpr size_t ALIGN256(size_t x) { return (x + 255) & ~size_t(255); }
constexpr size_t O_NORMED  = 0;                                              // f16 M x 128
constexpr size_t O_WINP    = ALIGN256(O_NORMED  + (size_t)MROWS*CH*2);       // packed B inproj
constexpr size_t O_ZXB     = ALIGN256(O_WINP    + (size_t)KT_INPROJ*NT_INPROJ*512*2);
constexpr size_t O_XBC     = ALIGN256(O_ZXB     + (size_t)MROWS*NPROJ_PAD*4);// f32 M x 384 (post conv+silu)
constexpr size_t O_DT      = ALIGN256(O_XBC     + (size_t)MROWS*CONVCH*4);
constexpr size_t O_DA      = ALIGN256(O_DT      + (size_t)MROWS*NHEADS*4);
constexpr size_t O_Y       = ALIGN256(O_DA      + (size_t)MROWS*NHEADS*4);   // f32 M x 256 (scan out)
constexpr size_t O_YF16    = ALIGN256(O_Y       + (size_t)MROWS*DINNER*4);   // f16 M x 256
constexpr size_t O_WOUT    = ALIGN256(O_YF16    + (size_t)MROWS*DINNER*2);
constexpr size_t O_M       = ALIGN256(O_WOUT    + (size_t)KT_OUT*NT_OUT*512*2); // f32 M x 128
constexpr size_t O_Y1      = ALIGN256(O_M       + (size_t)MROWS*CH*4);       // f32 NCHW residual
constexpr size_t O_Y1H     = ALIGN256(O_Y1      + (size_t)MROWS*CH*4);       // f16 NHWC conv input
constexpr size_t O_IM2COL  = ALIGN256(O_Y1H     + (size_t)MROWS*CH*2);       // f16 M x 1152
constexpr size_t O_WFF1    = ALIGN256(O_IM2COL  + (size_t)MROWS*KIM*2);
constexpr size_t O_WFF2    = ALIGN256(O_WFF1    + (size_t)KT_FF*NT_FF*512*2);
constexpr size_t O_FF      = ALIGN256(O_WFF2    + (size_t)KT_FF*NT_FF*512*2);// f32 M x 128 (gemm out)
constexpr size_t O_FFH     = ALIGN256(O_FF      + (size_t)MROWS*CH*4);       // f16 M x 128 (relu'd)

__device__ __forceinline__ float siluf(float x) { return x / (1.0f + __expf(-x)); }
__device__ __forceinline__ int reflect64(int v) {
  if (v < 0) v = -v;
  if (v >= IMG) v = 2*IMG - 2 - v;
  return v;
}

// ---------------- LayerNorm -> f16 -------------------
__global__ void layernorm_f16_kernel(const float* __restrict__ noisy,
                                     const float* __restrict__ gamma,
                                     const float* __restrict__ beta,
                                     _Float16* __restrict__ out) {
  int row = blockIdx.x;        // b*SEQ + t
  int c   = threadIdx.x;       // 0..127
  int b = row / SEQ, t = row % SEQ;
  float v = noisy[((size_t)b*CH + c)*SEQ + t];
  __shared__ float s1[CH], s2[CH];
  s1[c] = v; s2[c] = v*v;
  __syncthreads();
  for (int off = CH/2; off > 0; off >>= 1) {
    if (c < off) { s1[c] += s1[c+off]; s2[c] += s2[c+off]; }
    __syncthreads();
  }
  float mu  = s1[0] * (1.0f/CH);
  float var = s2[0] * (1.0f/CH) - mu*mu;
  float nv  = (v - mu) * rsqrtf(var + 1e-5f) * gamma[c] + beta[c];
  out[(size_t)row*CH + c] = (_Float16)nv;
}

// ------- pack row-major W (Nreal x K) into WMMA B fragments -------
// B 32x16 f16 fragment, contiguous k order (same as A): lane<16 -> N=lane,  K = kt*32+0..15
//                                            lane>=16 -> N=lane-16, K = kt*32+16..31
__global__ void pack_b_kernel(const float* __restrict__ W, _Float16* __restrict__ Bp,
                              int Nreal, int K, int ntiles, int ktiles) {
  int gid = blockIdx.x * blockDim.x + threadIdx.x;
  int total = ktiles * ntiles * 32;
  if (gid >= total) return;
  int lane = gid & 31;
  int frag = gid >> 5;
  int nt = frag % ntiles;
  int kt = frag / ntiles;
  int n = nt*16 + (lane & 15);
  int kbase = kt*32 + ((lane >> 4) * 16);
  _Float16* dst = Bp + (size_t)frag*512 + lane*16;
  #pragma unroll
  for (int m = 0; m < 16; ++m) {
    int k = kbase + m;
    float v = (n < Nreal && k < K) ? W[(size_t)n*K + k] : 0.0f;
    dst[m] = (_Float16)v;
  }
}

// ------- generic WMMA GEMM: C(MxNpad,f32) = A(MxK,f16,row-major) * Bpacked -------
// 8 waves/block; wave w owns M-tile blockIdx.y*8+w; block owns 4 consecutive N-tiles.
// B fragments (shared by all 8 waves) are staged into LDS with the CDNA5 async
// LDS engine (GLOBAL_LOAD_ASYNC_TO_LDS_B128, ASYNCcnt), ping/pong buffered.
// ktiles must be even and >= 2: steady-state loop processes 2 K-tiles branch-free.
__global__ __launch_bounds__(256) void wmma_gemm_kernel(
    const _Float16* __restrict__ A, const _Float16* __restrict__ Bp,
    float* __restrict__ C, int lda, int ktiles, int ntiles) {
  __shared__ __align__(16) _Float16 sB[2][4 * 512];   // 2 x 4KB (4 B-fragments each)

  int tid   = threadIdx.x;
  int lane  = tid & 31;
  int wave  = tid >> 5;
  int mtile = blockIdx.y * 8 + wave;
  int ntb   = blockIdx.x * 4;
  int mrow  = mtile*16 + (lane & 15);
  int kgrp  = (lane >> 4);                // A: lanes>=16 start at K+16
  const _Float16* arow = A + (size_t)mrow*lda + kgrp*16;

  v8f acc[4];
  #pragma unroll
  for (int nt = 0; nt < 4; ++nt)
    #pragma unroll
    for (int r = 0; r < 8; ++r) acc[nt][r] = 0.0f;

  // async-stage the contiguous 4KB B chunk for k-tile kt into sB[buf]
  auto stageB = [&](int kt, int buf) {
    const _Float16* g = Bp + ((size_t)kt*ntiles + ntb)*512 + tid*8;  // 16B per thread
    unsigned ldsoff = (unsigned)(uintptr_t)(&sB[buf][tid*8]);
    asm volatile("global_load_async_to_lds_b128 %0, %1, off"
                 :: "v"(ldsoff), "v"((unsigned long long)(uintptr_t)g)
                 : "memory");
  };
  auto loadA = [&](int kt, v8h& lo, v8h& hi) {
    lo = *(const v8h*)(arow + (size_t)kt*32);
    hi = *(const v8h*)(arow + (size_t)kt*32 + 8);
  };
  auto compute = [&](int buf, const v8h& lo, const v8h& hi) {
    v16h a;
    #pragma unroll
    for (int i = 0; i < 8; ++i) { a[i] = lo[i]; a[8+i] = hi[i]; }
    #pragma unroll
    for (int nt = 0; nt < 4; ++nt) {
      const _Float16* bl = &sB[buf][nt*512 + lane*16];
      v8h blo = *(const v8h*)bl;                      // ds_load_b128
      v8h bhi = *(const v8h*)(bl + 8);
      v16h bf;
      #pragma unroll
      for (int i = 0; i < 8; ++i) { bf[i] = blo[i]; bf[8+i] = bhi[i]; }
      acc[nt] = __builtin_amdgcn_wmma_f32_16x16x32_f16(
          false, a, false, bf, (short)0, acc[nt], false, false);
    }
  };

  v8h a_lo, a_hi, b_lo, b_hi;
  stageB(0, 0);
  loadA(0, a_lo, a_hi);

  int kt = 0;
  // branch-free steady state: two K-tiles per trip, static ping/pong buffers
  for (; kt < ktiles - 2; kt += 2) {
    stageB(kt + 1, 1);
    loadA(kt + 1, b_lo, b_hi);
    __builtin_prefetch(arow + (size_t)(kt+3)*32, 0, 1);
    asm volatile("s_wait_asynccnt 0x1" ::: "memory");  // stage(kt) done
    __syncthreads();
    compute(0, a_lo, a_hi);
    __syncthreads();                                   // buf0 free for re-stage

    stageB(kt + 2, 0);
    loadA(kt + 2, a_lo, a_hi);
    asm volatile("s_wait_asynccnt 0x1" ::: "memory");  // stage(kt+1) done
    __syncthreads();
    compute(1, b_lo, b_hi);
    __syncthreads();                                   // buf1 free for re-stage
  }

  // epilogue: last two K-tiles (kt == ktiles-2)
  stageB(kt + 1, 1);
  loadA(kt + 1, b_lo, b_hi);
  asm volatile("s_wait_asynccnt 0x1" ::: "memory");
  __syncthreads();
  compute(0, a_lo, a_hi);
  asm volatile("s_wait_asynccnt 0x0" ::: "memory");    // stage(kt+1) done
  __syncthreads();
  compute(1, b_lo, b_hi);

  // C/D layout: VGPR r -> row r (lanes 0-15) / row 8+r (lanes 16-31); N = lane%16
  int npad   = ntiles * 16;
  int ncol   = lane & 15;
  int rowoff = (lane >> 4) * 8;
  #pragma unroll
  for (int nt = 0; nt < 4; ++nt) {
    int n = (ntb + nt)*16 + ncol;
    #pragma unroll
    for (int r = 0; r < 8; ++r)
      C[(size_t)(mtile*16 + rowoff + r)*npad + n] = acc[nt][r];
  }
}

// ------- depthwise causal conv1d(k=5) + bias + SiLU -------
__global__ void conv1d_silu_kernel(const float* __restrict__ zxb,
                                   const float* __restrict__ cw,
                                   const float* __restrict__ cb,
                                   float* __restrict__ xbc) {
  size_t gid = (size_t)blockIdx.x * blockDim.x + threadIdx.x;
  if (gid >= (size_t)MROWS * CONVCH) return;
  int r  = (int)(gid / CONVCH);
  int ch = (int)(gid % CONVCH);
  int b = r / SEQ, t = r % SEQ;
  float s = 0.0f;
  #pragma unroll
  for (int j = 0; j < 5; ++j) {
    int tt = t - 4 + j;
    if (tt >= 0)
      s += cw[ch*5 + j] * zxb[((size_t)b*SEQ + tt)*NPROJ_PAD + DINNER + ch];
  }
  xbc[(size_t)r*CONVCH + ch] = siluf(s + cb[ch]);
}

// ------- dt = softplus(raw + bias); dA = exp(dt * -exp(A_log)) -------
__global__ void dtda_kernel(const float* __restrict__ zxb,
                            const float* __restrict__ dt_bias,
                            const float* __restrict__ A_log,
                            float* __restrict__ dt, float* __restrict__ dA) {
  int gid = blockIdx.x * blockDim.x + threadIdx.x;
  if (gid >= MROWS * NHEADS) return;
  int r = gid / NHEADS, h = gid % NHEADS;
  float v = zxb[(size_t)r*NPROJ_PAD + (2*DINNER + 2*DSTATE) + h] + dt_bias[h];
  float sp = (v > 20.0f) ? v : log1pf(__expf(v));
  dt[gid] = sp;
  dA[gid] = __expf(sp * (-__expf(A_log[h])));
}

// ------- sequential SSM scan: one block per (batch, head) -------
// 256 threads: lane owns p = tid/4 (head dim) and a 16-wide slice of the 64 states.
__global__ __launch_bounds__(256) void ssm_scan_kernel(
    const float* __restrict__ xbc, const float* __restrict__ dt,
    const float* __restrict__ dA, const float* __restrict__ Dskip,
    float* __restrict__ y) {
  int bb = blockIdx.x / NHEADS;
  int hh = blockIdx.x % NHEADS;
  int tid = threadIdx.x;
  int p  = tid >> 2;       // 0..63
  int ng = tid & 3;        // state-slice group
  float Dh = Dskip[hh];

  float h[16];
  #pragma unroll
  for (int i = 0; i < 16; ++i) h[i] = 0.0f;

  __shared__ float sx[DSTATE], sB[DSTATE], sC[DSTATE], ssc[2];

  for (int t = 0; t < SEQ; ++t) {
    size_t row = (size_t)bb*SEQ + t;
    if (tid < 64)        sx[tid]       = xbc[row*CONVCH + hh*64 + tid];
    else if (tid < 128)  sB[tid - 64]  = xbc[row*CONVCH + DINNER + (tid - 64)];
    else if (tid < 192)  sC[tid - 128] = xbc[row*CONVCH + DINNER + DSTATE + (tid - 128)];
    else if (tid == 192) ssc[0]        = dt[row*NHEADS + hh];
    else if (tid == 193) ssc[1]        = dA[row*NHEADS + hh];
    __syncthreads();

    float xt   = sx[p];
    float coef = ssc[0] * xt;
    float da   = ssc[1];
    float acc  = 0.0f;
    #pragma unroll
    for (int i = 0; i < 16; ++i) {
      int n = ng*16 + i;
      h[i] = h[i]*da + coef*sB[n];
      acc += h[i]*sC[n];
    }
    // reduce over the 4 lanes sharing p (wave32 shuffle)
    acc += __shfl_xor(acc, 1, 4);
    acc += __shfl_xor(acc, 2, 4);
    if (ng == 0) y[row*DINNER + hh*64 + p] = acc + xt*Dh;
    __syncthreads();
  }
}

// ------- y = y*silu(z); RMSNorm(256)*rms_w -> f16 -------
__global__ void gate_rms_kernel(const float* __restrict__ zxb,
                                const float* __restrict__ yin,
                                const float* __restrict__ rms_w,
                                _Float16* __restrict__ yout) {
  int row = blockIdx.x;
  int c   = threadIdx.x;   // 0..255
  float z = zxb[(size_t)row*NPROJ_PAD + c];
  float g = yin[(size_t)row*DINNER + c] * siluf(z);
  __shared__ float s[DINNER];
  s[c] = g*g;
  __syncthreads();
  for (int off = DINNER/2; off > 0; off >>= 1) {
    if (c < off) s[c] += s[c+off];
    __syncthreads();
  }
  float scale = rsqrtf(s[0] * (1.0f/DINNER) + 1e-5f);
  yout[(size_t)row*DINNER + c] = (_Float16)(g * scale * rms_w[c]);
}

// ------- y1 = noisy + m (NCHW) ; also NHWC f16 copy for conv -------
__global__ void residual_kernel(const float* __restrict__ noisy,
                                const float* __restrict__ m,
                                float* __restrict__ y1,
                                _Float16* __restrict__ y1h) {
  size_t gid = (size_t)blockIdx.x * blockDim.x + threadIdx.x;
  if (gid >= (size_t)MROWS * CH) return;
  int r = (int)(gid / CH);
  int c = (int)(gid % CH);
  int b = r / SEQ, t = r % SEQ;
  size_t nchw = ((size_t)b*CH + c)*SEQ + t;
  float v = noisy[nchw] + m[gid];
  y1[nchw]  = v;
  y1h[gid]  = (_Float16)v;
}

// ------- im2col with reflect padding (NHWC f16 -> M x 1152 f16) -------
__global__ void im2col_kernel(const _Float16* __restrict__ src,
                              _Float16* __restrict__ dst) {
  size_t gid = (size_t)blockIdx.x * blockDim.x + threadIdx.x;
  if (gid >= (size_t)MROWS * KIM) return;
  int r = (int)(gid / KIM);
  int k = (int)(gid % KIM);
  int cin = k / 9, rem = k % 9, ky = rem / 3, kx = rem % 3;
  int b = r / SEQ, t = r % SEQ;
  int yy = t / IMG, xx = t % IMG;
  int ry = reflect64(yy + ky - 1);
  int rx = reflect64(xx + kx - 1);
  dst[gid] = src[((size_t)b*SEQ + ry*IMG + rx)*CH + cin];
}

// ------- relu(x + bias) -> f16 -------
__global__ void relu_bias_f16_kernel(const float* __restrict__ x,
                                     const float* __restrict__ bias,
                                     _Float16* __restrict__ out) {
  size_t gid = (size_t)blockIdx.x * blockDim.x + threadIdx.x;
  if (gid >= (size_t)MROWS * CH) return;
  int c = (int)(gid % CH);
  float v = x[gid] + bias[c];
  out[gid] = (_Float16)(v > 0.0f ? v : 0.0f);
}

// ------- final: out_y = y1 + relu(ff2 + b2); out_aux = aux -------
__global__ void final_kernel(const float* __restrict__ y1,
                             const float* __restrict__ ff,
                             const float* __restrict__ b2,
                             const float* __restrict__ aux,
                             float* __restrict__ out) {
  size_t gid = (size_t)blockIdx.x * blockDim.x + threadIdx.x;
  constexpr size_t NT = (size_t)MROWS * CH;
  if (gid >= NT) return;
  int b = (int)(gid / ((size_t)CH * SEQ));
  int c = (int)((gid / SEQ) % CH);
  int t = (int)(gid % SEQ);
  float f = ff[((size_t)b*SEQ + t)*CH + c] + b2[c];
  f = f > 0.0f ? f : 0.0f;
  out[gid]      = y1[gid] + f;
  out[NT + gid] = aux[gid];
}

// =====================================================
extern "C" void kernel_launch(void* const* d_in, const int* in_sizes, int n_in,
                              void* d_out, int out_size, void* d_ws, size_t ws_size,
                              hipStream_t stream) {
  (void)in_sizes; (void)n_in; (void)out_size; (void)ws_size;
  const float* noisy      = (const float*)d_in[0];
  const float* aux        = (const float*)d_in[1];
  const float* ln_gamma   = (const float*)d_in[2];
  const float* ln_beta    = (const float*)d_in[3];
  const float* in_proj_w  = (const float*)d_in[4];
  const float* conv_w     = (const float*)d_in[5];
  const float* conv_b     = (const float*)d_in[6];
  const float* A_log      = (const float*)d_in[7];
  const float* dt_bias    = (const float*)d_in[8];
  const float* Dskip      = (const float*)d_in[9];
  const float* rms_w      = (const float*)d_in[10];
  const float* out_proj_w = (const float*)d_in[11];
  const float* ff_w1      = (const float*)d_in[12];
  const float* ff_b1      = (const float*)d_in[13];
  const float* ff_w2      = (const float*)d_in[14];
  const float* ff_b2      = (const float*)d_in[15];
  float* out = (float*)d_out;

  char* ws = (char*)d_ws;
  _Float16* normed  = (_Float16*)(ws + O_NORMED);
  _Float16* winp    = (_Float16*)(ws + O_WINP);
  float*    zxb     = (float*)   (ws + O_ZXB);
  float*    xbc     = (float*)   (ws + O_XBC);
  float*    dt      = (float*)   (ws + O_DT);
  float*    dA      = (float*)   (ws + O_DA);
  float*    yscan   = (float*)   (ws + O_Y);
  _Float16* yf16    = (_Float16*)(ws + O_YF16);
  _Float16* wout    = (_Float16*)(ws + O_WOUT);
  float*    mproj   = (float*)   (ws + O_M);
  float*    y1      = (float*)   (ws + O_Y1);
  _Float16* y1h     = (_Float16*)(ws + O_Y1H);
  _Float16* im2c    = (_Float16*)(ws + O_IM2COL);
  _Float16* wff1    = (_Float16*)(ws + O_WFF1);
  _Float16* wff2    = (_Float16*)(ws + O_WFF2);
  float*    ff      = (float*)   (ws + O_FF);
  _Float16* ffh     = (_Float16*)(ws + O_FFH);

  auto cdiv = [](size_t a, size_t b) { return (unsigned)((a + b - 1) / b); };

  // 0) weight packing (independent)
  pack_b_kernel<<<cdiv((size_t)KT_INPROJ*NT_INPROJ*32, 256), 256, 0, stream>>>(
      in_proj_w, winp, NPROJ, CH, NT_INPROJ, KT_INPROJ);
  pack_b_kernel<<<cdiv((size_t)KT_OUT*NT_OUT*32, 256), 256, 0, stream>>>(
      out_proj_w, wout, CH, DINNER, NT_OUT, KT_OUT);
  pack_b_kernel<<<cdiv((size_t)KT_FF*NT_FF*32, 256), 256, 0, stream>>>(
      ff_w1, wff1, CH, KIM, NT_FF, KT_FF);
  pack_b_kernel<<<cdiv((size_t)KT_FF*NT_FF*32, 256), 256, 0, stream>>>(
      ff_w2, wff2, CH, KIM, NT_FF, KT_FF);

  // 1) LayerNorm -> f16
  layernorm_f16_kernel<<<MROWS, CH, 0, stream>>>(noisy, ln_gamma, ln_beta, normed);

  // 2) in-proj GEMM (M=16384, K=128, Npad=704)
  wmma_gemm_kernel<<<dim3(NT_INPROJ/4, MROWS/128), 256, 0, stream>>>(
      normed, winp, zxb, CH, KT_INPROJ, NT_INPROJ);

  // 3) depthwise causal conv1d + SiLU ; dt/dA
  conv1d_silu_kernel<<<cdiv((size_t)MROWS*CONVCH, 256), 256, 0, stream>>>(
      zxb, conv_w, conv_b, xbc);
  dtda_kernel<<<cdiv((size_t)MROWS*NHEADS, 256), 256, 0, stream>>>(
      zxb, dt_bias, A_log, dt, dA);

  // 4) sequential SSM scan (one block per batch*head)
  ssm_scan_kernel<<<BATCH*NHEADS, 256, 0, stream>>>(xbc, dt, dA, Dskip, yscan);

  // 5) gate + RMSNorm -> f16
  gate_rms_kernel<<<MROWS, DINNER, 0, stream>>>(zxb, yscan, rms_w, yf16);

  // 6) out-proj GEMM (M=16384, K=256, N=128)
  wmma_gemm_kernel<<<dim3(NT_OUT/4, MROWS/128), 256, 0, stream>>>(
      yf16, wout, mproj, DINNER, KT_OUT, NT_OUT);

  // 7) residual add -> y1 (NCHW f32) + NHWC f16
  residual_kernel<<<cdiv((size_t)MROWS*CH, 256), 256, 0, stream>>>(
      noisy, mproj, y1, y1h);

  // 8) conv1: im2col -> GEMM -> relu+bias -> f16
  im2col_kernel<<<cdiv((size_t)MROWS*KIM, 256), 256, 0, stream>>>(y1h, im2c);
  wmma_gemm_kernel<<<dim3(NT_FF/4, MROWS/128), 256, 0, stream>>>(
      im2c, wff1, ff, KIM, KT_FF, NT_FF);
  relu_bias_f16_kernel<<<cdiv((size_t)MROWS*CH, 256), 256, 0, stream>>>(ff, ff_b1, ffh);

  // 9) conv2: im2col -> GEMM
  im2col_kernel<<<cdiv((size_t)MROWS*KIM, 256), 256, 0, stream>>>(ffh, im2c);
  wmma_gemm_kernel<<<dim3(NT_FF/4, MROWS/128), 256, 0, stream>>>(
      im2c, wff2, ff, KIM, KT_FF, NT_FF);

  // 10) final: y = y1 + relu(ff + b2); aux passthrough
  final_kernel<<<cdiv((size_t)MROWS*CH, 256), 256, 0, stream>>>(
      y1, ff, ff_b2, aux, out);
}
